// SocialLSTM_Downsample_62010737820027
// MI455X (gfx1250) — compile-verified, matrix operand order checked by reading, not run on hardware
//
#include <hip/hip_runtime.h>
#include <hip/hip_bf16.h>
#include <math.h>

// ---------------------------------------------------------------------------
// Social-LSTM on gfx1250. All GEMMs via v_wmma_f32_16x16x32_f16 (wave32).
// Weight B-fragments staged once per block into dynamic LDS (CDNA5: 320KB/WGP).
// ---------------------------------------------------------------------------

typedef __attribute__((ext_vector_type(16))) _Float16 v16h;
typedef __attribute__((ext_vector_type(8)))  _Float16 v8h;
typedef __attribute__((ext_vector_type(8)))  float    v8f;

#define GSZ   3
#define NCELL 9
#define EMB   32
#define HID   32
#define TSEQ  64
#define CH    2
#define STEP  5
#define BATCH 4096
#define BT    16                 // batch tile (M of WMMA)
#define NBT   (BATCH / BT)       // 256 batch tiles
#define FRAG  512                // halves per fragment (32 lanes x 16 halves)

// workspace layout, in _Float16 elements (all fragment-aligned: x512)
#define WIH_OFF 0                                   // [cell][kt:2][nt:8] -> 144 frags
#define WHH_OFF (WIH_OFF + NCELL * 2 * 8 * FRAG)    // [cell][nt:8]      ->  72 frags
#define SOC_OFF (WHH_OFF + NCELL * 8 * FRAG)        // [kt:9][nt:2]      ->  18 frags
#define NWFRAG  (NCELL * 2 * 8 + NCELL * 8 + NCELL * 2)   // 234 fragments
#define FRM_OFF (SOC_OFF + NCELL * 2 * FRAG)        // [t][btile][cell]  A-frags

// dynamic-LDS layout (bytes)
#define L_WB    (NWFRAG * FRAG * 2)                 // 239616 : weight fragments
#define L_H     (L_WB)                              // [9][16][32] f16 : 9216
#define L_SOC   (L_H + NCELL * BT * HID * 2)        // [16][32]  f16 : 1024
#define L_PART  (L_SOC + BT * EMB * 2)              // [9][16][32] f32 : 18432
#define L_TOTAL (L_PART + NCELL * BT * EMB * 4)     // 268288 bytes

static __device__ __forceinline__ v8f wmma_f16(v16h a, v16h b, v8f c) {
  return __builtin_amdgcn_wmma_f32_16x16x32_f16(
      /*neg_a=*/false, a, /*neg_b=*/false, b,
      /*c_mod=*/(short)0, c, /*reuse_a=*/false, /*reuse_b=*/false);
}

static __device__ __forceinline__ float sigmoidf_(float x) {
  return 1.0f / (1.0f + __expf(-x));
}

// ---------------------------------------------------------------------------
// Kernel 1: downsample (mean/max/min over 4x4 per grid cell) + embed + ReLU,
// stored pre-swizzled as WMMA A-fragments (M=16 batches, K=32 embeddings).
// One wave per fragment (t, btile, cell). Each half-wave pools one channel;
// stats exchanged with shfl_xor(16) so x is read exactly once (~302MB).
// ---------------------------------------------------------------------------
__global__ void k_frames(const float* __restrict__ x,
                         const float* __restrict__ down_W,
                         const float* __restrict__ down_b,
                         _Float16*    __restrict__ frm) {
  const int wavesPerBlk = blockDim.x >> 5;
  const int fragId = blockIdx.x * wavesPerBlk + (threadIdx.x >> 5);
  const int nFrag  = TSEQ * NBT * NCELL;
  if (fragId >= nFrag) return;                       // uniform per wave

  const int lane = threadIdx.x & 31;
  const int cell = fragId % NCELL;
  const int bt   = (fragId / NCELL) % NBT;
  const int t    = fragId / (NCELL * NBT);
  const int gy   = cell / GSZ, gx = cell % GSZ;
  const int m    = lane & 15;
  const int b    = bt * BT + m;
  const int g16  = lane >> 4;                        // half-wave id == channel
  const int eg   = g16 * 8;                          // K-group base per half-wave

  // pool the 4x4 block of my channel (g16)
  const float* px = x + (((size_t)(b * CH + g16) * TSEQ + t) * 144
                         + gy * 4 * 12 + gx * 4);
  float s = 0.f, mx = -1e30f, mn = 1e30f;
  for (int yy = 0; yy < 4; ++yy)
    for (int xx = 0; xx < 4; ++xx) {
      float v = px[yy * 12 + xx];
      s += v; mx = fmaxf(mx, v); mn = fminf(mn, v);
    }
  float me = s * (1.0f / 16.0f);
  float o_me = __shfl_xor(me, 16);
  float o_mx = __shfl_xor(mx, 16);
  float o_mn = __shfl_xor(mn, 16);

  float pooled[6];                                   // [stat*2 + ch]
  pooled[0 + g16] = me;  pooled[0 + (1 - g16)] = o_me;
  pooled[2 + g16] = mx;  pooled[2 + (1 - g16)] = o_mx;
  pooled[4 + g16] = mn;  pooled[4 + (1 - g16)] = o_mn;

  v16h out;
  for (int i = 0; i < 16; ++i) {
    int e = (i < 8) ? (eg + i) : (eg + 16 + (i - 8));
    float v = down_b[e];
    for (int c2 = 0; c2 < 6; ++c2) v += pooled[c2] * down_W[e * 6 + c2];
    out[i] = (_Float16)fmaxf(v, 0.0f);
  }
  *(v16h*)(frm + (size_t)fragId * FRAG + lane * 16) = out;
}

// ---------------------------------------------------------------------------
// Kernel 2: swizzle weights into WMMA B-fragment layout (K x N, f16).
// One wave per fragment. 144 (W_ih) + 72 (W_hh) + 18 (social_W) = 234 frags.
// ---------------------------------------------------------------------------
__global__ void k_weights(const float* __restrict__ W_ih,   // (9,128,64)
                          const float* __restrict__ W_hh,   // (9,128,32)
                          const float* __restrict__ soc_W,  // (32,288)
                          _Float16*    __restrict__ ws) {
  const int wavesPerBlk = blockDim.x >> 5;
  const int f = blockIdx.x * wavesPerBlk + (threadIdx.x >> 5);
  if (f >= NWFRAG) return;
  const int lane = threadIdx.x & 31;
  const int n  = lane & 15;                // column of B tile
  const int kg = (lane >> 4) * 8;          // K-group base

  v16h out;
  _Float16* dst;
  if (f < 144) {                           // W_ih : [cell][kt][nt]
    const int cell = f / 16, kt = (f / 8) & 1, nt = f & 7;
    for (int i = 0; i < 16; ++i) {
      int kl = (i < 8) ? (kg + i) : (kg + 16 + (i - 8));
      out[i] = (_Float16)W_ih[((size_t)cell * 128 + nt * 16 + n) * 64 + kt * 32 + kl];
    }
    dst = ws + WIH_OFF + (size_t)f * FRAG;
  } else if (f < 216) {                    // W_hh : [cell][nt]
    const int g = f - 144, cell = g / 8, nt = g & 7;
    for (int i = 0; i < 16; ++i) {
      int kl = (i < 8) ? (kg + i) : (kg + 16 + (i - 8));
      out[i] = (_Float16)W_hh[((size_t)cell * 128 + nt * 16 + n) * 32 + kl];
    }
    dst = ws + WHH_OFF + (size_t)g * FRAG;
  } else {                                 // social_W : [kt=cell][nt]
    const int g = f - 216, kt = g / 2, nt = g & 1;
    for (int i = 0; i < 16; ++i) {
      int kl = (i < 8) ? (kg + i) : (kg + 16 + (i - 8));
      out[i] = (_Float16)soc_W[(size_t)(nt * 16 + n) * 288 + kt * 32 + kl];
    }
    dst = ws + SOC_OFF + (size_t)g * FRAG;
  }
  *(v16h*)(dst + lane * 16) = out;
}

// ---------------------------------------------------------------------------
// Kernel 3: recurrence. One block = one 16-batch tile, 9 waves (1 per cell).
// All 234KB of weight B-fragments staged in dynamic LDS once per block.
// Social GEMM split across waves: each wave computes its own cell's K-partial
// (2 WMMAs), then a 288-thread reduction combines the 9 partials.
// Conv over t folded into register accumulators; final out_W dot via shfl_xor.
// ---------------------------------------------------------------------------
__global__ void __launch_bounds__(288)
k_lstm(const _Float16* __restrict__ ws,      // weight + frame fragments
       const float* __restrict__ b_ih, const float* __restrict__ b_hh,
       const float* __restrict__ soc_b,
       const float* __restrict__ conv_W, const float* __restrict__ conv_b,
       const float* __restrict__ out_W,  const float* __restrict__ out_b,
       float* __restrict__ out) {
  extern __shared__ __align__(16) char smem[];
  _Float16* wfrag   = (_Float16*)(smem);            // 234 B-fragments
  _Float16* h_lds   = (_Float16*)(smem + L_H);      // [9][16][32]
  _Float16* soc_lds = (_Float16*)(smem + L_SOC);    // [16][32]
  float*    part    = (float*)   (smem + L_PART);   // [9][16][32]

  const int tid  = threadIdx.x;
  const int w    = tid >> 5;               // wave id == cell
  const int lane = tid & 31;
  const int cell = w;
  const int bt   = blockIdx.x;
  const int n    = lane & 15;
  const int kg   = (lane >> 4) * 8;
  const _Float16* frm = ws + FRM_OFF;

  // ---- stage all weight fragments into LDS (one time, ~234KB) ----
  {
    const uint4* src = (const uint4*)ws;
    uint4*       dst = (uint4*)smem;
    for (int i = tid; i < L_WB / 16; i += 288) dst[i] = src[i];
  }
  // zero h state in LDS
  for (int i = tid; i < NCELL * BT * HID; i += 288) h_lds[i] = (_Float16)0.0f;

  // hoisted per-lane constants
  float gbias[8];
  for (int nt = 0; nt < 8; ++nt) {
    int k = nt * 16 + n;
    gbias[nt] = b_ih[cell * 128 + k] + b_hh[cell * 128 + k];
  }

  v8f c_state[2], accv[STEP][2];
  for (int jt = 0; jt < 2; ++jt)
    for (int q = 0; q < 8; ++q) c_state[jt][q] = 0.0f;
  for (int s = 0; s < STEP; ++s)
    for (int jt = 0; jt < 2; ++jt)
      for (int q = 0; q < 8; ++q) accv[s][jt][q] = 0.0f;

  __syncthreads();

  for (int t = 0; t < TSEQ; ++t) {
    // ---- assemble A fragment of my own cell's h (written by this wave) ----
    const _Float16* hrow = h_lds + (cell * BT + n) * HID;
    v8h hh0 = *(const v8h*)(hrow + kg);
    v8h hh1 = *(const v8h*)(hrow + kg + 16);
    v16h ah = __builtin_shufflevector(hh0, hh1, 0, 1, 2, 3, 4, 5, 6, 7,
                                      8, 9, 10, 11, 12, 13, 14, 15);

    // ---- social K-partial for my cell: (16x32) += h_cell(16x32)@W(32x32) ----
    v8f sp[2];
    for (int nt = 0; nt < 2; ++nt)
      for (int q = 0; q < 8; ++q) sp[nt][q] = 0.0f;
    for (int nt = 0; nt < 2; ++nt) {
      v16h bm = *(const v16h*)(wfrag + SOC_OFF + (size_t)(cell * 2 + nt) * FRAG
                               + lane * 16);
      sp[nt] = wmma_f16(ah, bm, sp[nt]);
    }
    for (int nt = 0; nt < 2; ++nt)
      for (int r = 0; r < 8; ++r)
        part[(cell * BT + (r + 8 * (lane >> 4))) * EMB + n + nt * 16] = sp[nt][r];
    __syncthreads();

    // ---- reduce 9 partials -> social = relu(sum + soc_b) ----
    for (int idx = tid; idx < BT * EMB; idx += 288) {
      int e = idx & 31;
      float v = soc_b[e];
      for (int kt = 0; kt < NCELL; ++kt) v += part[kt * BT * EMB + idx];
      soc_lds[idx] = (_Float16)fmaxf(v, 0.0f);
    }
    __syncthreads();

    // ---- A fragments: frame (precomputed), social ----
    v16h af = *(const v16h*)(frm + ((size_t)(t * NBT + bt) * NCELL + cell) * FRAG
                             + lane * 16);
    v8h s0 = *(const v8h*)(soc_lds + n * EMB + kg);
    v8h s1 = *(const v8h*)(soc_lds + n * EMB + kg + 16);
    v16h as = __builtin_shufflevector(s0, s1, 0, 1, 2, 3, 4, 5, 6, 7,
                                      8, 9, 10, 11, 12, 13, 14, 15);

    // ---- gate GEMM, A-major (8 consecutive WMMAs share the same A) ----
    v8f gacc[8];
    for (int nt = 0; nt < 8; ++nt)
      for (int q = 0; q < 8; ++q) gacc[nt][q] = gbias[nt];
    for (int nt = 0; nt < 8; ++nt) {
      v16h bm = *(const v16h*)(wfrag + WIH_OFF
                               + (size_t)((cell * 2 + 0) * 8 + nt) * FRAG + lane * 16);
      gacc[nt] = wmma_f16(af, bm, gacc[nt]);
    }
    for (int nt = 0; nt < 8; ++nt) {
      v16h bm = *(const v16h*)(wfrag + WIH_OFF
                               + (size_t)((cell * 2 + 1) * 8 + nt) * FRAG + lane * 16);
      gacc[nt] = wmma_f16(as, bm, gacc[nt]);
    }
    for (int nt = 0; nt < 8; ++nt) {
      v16h bm = *(const v16h*)(wfrag + WHH_OFF
                               + (size_t)(cell * 8 + nt) * FRAG + lane * 16);
      gacc[nt] = wmma_f16(ah, bm, gacc[nt]);
    }

    // ---- LSTM nonlinearities on D fragments (i,f,g,o = nt 0-1,2-3,4-5,6-7) --
    float hbuf[2][8];
    for (int jt = 0; jt < 2; ++jt)
      for (int q = 0; q < 8; ++q) {
        float ig = sigmoidf_(gacc[0 + jt][q]);
        float fg = sigmoidf_(gacc[2 + jt][q]);
        float gg = tanhf(gacc[4 + jt][q]);
        float og = sigmoidf_(gacc[6 + jt][q]);
        float cn = fg * c_state[jt][q] + ig * gg;
        c_state[jt][q] = cn;
        hbuf[jt][q] = og * tanhf(cn);
      }

    // ---- fold conv3d (1x1x1 over t) into register accumulators ----
    for (int s = 0; s < STEP; ++s) {
      float cw = conv_W[s * TSEQ + t];
      for (int jt = 0; jt < 2; ++jt)
        for (int q = 0; q < 8; ++q) accv[s][jt][q] += cw * hbuf[jt][q];
    }

    // ---- publish h_t (only this wave reads it back) ----
    for (int jt = 0; jt < 2; ++jt)
      for (int q = 0; q < 8; ++q)
        h_lds[(cell * BT + (q + 8 * (lane >> 4))) * HID + n + 16 * jt] =
            (_Float16)hbuf[jt][q];

    // prefetch next frame fragment (global_prefetch_b8)
    if (t + 1 < TSEQ)
      __builtin_prefetch(frm + ((size_t)((t + 1) * NBT + bt) * NCELL + cell) * FRAG
                         + lane * 16, 0, 0);
    __syncthreads();
  }

  // ---- fused = relu(conv_acc + conv_b); out = fused @ out_W.T + out_b ----
  const float ow0 = out_W[n], ow1 = out_W[16 + n], ob = out_b[0];
  for (int s = 0; s < STEP; ++s) {
    float cb = conv_b[s];
    for (int r = 0; r < 8; ++r) {
      float f0 = fmaxf(accv[s][0][r] + cb, 0.0f);
      float f1 = fmaxf(accv[s][1][r] + cb, 0.0f);
      float p = f0 * ow0 + f1 * ow1;
      p += __shfl_xor(p, 1);
      p += __shfl_xor(p, 2);
      p += __shfl_xor(p, 4);
      p += __shfl_xor(p, 8);                 // reduced within each 16-lane half
      if (n == 0) {
        int m = r + 8 * (lane >> 4);
        int b = bt * BT + m;
        out[((size_t)s * BATCH + b) * NCELL + cell] = p + ob;
      }
    }
  }
}

// ---------------------------------------------------------------------------
extern "C" void kernel_launch(void* const* d_in, const int* in_sizes, int n_in,
                              void* d_out, int out_size, void* d_ws, size_t ws_size,
                              hipStream_t stream) {
  const float* x      = (const float*)d_in[0];
  const float* down_W = (const float*)d_in[1];
  const float* down_b = (const float*)d_in[2];
  const float* soc_W  = (const float*)d_in[3];
  const float* soc_b  = (const float*)d_in[4];
  const float* W_ih   = (const float*)d_in[5];
  const float* W_hh   = (const float*)d_in[6];
  const float* b_ih   = (const float*)d_in[7];
  const float* b_hh   = (const float*)d_in[8];
  const float* conv_W = (const float*)d_in[9];
  const float* conv_b = (const float*)d_in[10];
  const float* out_W  = (const float*)d_in[11];
  const float* out_b  = (const float*)d_in[12];
  float* out = (float*)d_out;
  _Float16* ws = (_Float16*)d_ws;

  // allow 262KB dynamic LDS (CDNA5 WGP has 320KB)
  (void)hipFuncSetAttribute((const void*)k_lstm,
                            hipFuncAttributeMaxDynamicSharedMemorySize, L_TOTAL);

  // 1) weight swizzle: 234 fragments, 8 waves/block
  k_weights<<<(NWFRAG + 7) / 8, 256, 0, stream>>>(W_ih, W_hh, soc_W, ws);

  // 2) frame embed + swizzle: 64*256*9 fragments, 8 waves/block
  {
    int nFrag = TSEQ * NBT * NCELL;
    k_frames<<<(nFrag + 7) / 8, 256, 0, stream>>>(x, down_W, down_b, ws + FRM_OFF);
  }

  // 3) recurrence: one block per 16-batch tile, 9 waves (one per grid cell)
  k_lstm<<<NBT, 288, L_TOTAL, stream>>>(ws, b_ih, b_hh, soc_b,
                                        conv_W, conv_b, out_W, out_b, out);
}
